// MLA_32212254720207
// MI455X (gfx1250) — matrix operand order repaired and checked by reading, not compiled, not run on hardware
//
#include <hip/hip_runtime.h>

// ---------------------------------------------------------------------------
// MLA forward for MI455X (gfx1250, wave32, WMMA).
// bf16 operands end-to-end; GEMMs stage tiles with GLOBAL_LOAD_ASYNC_TO_LDS
// (ASYNCcnt double-buffer), attention stages K/V tiles with the Tensor Data
// Mover (TENSOR_LOAD_TO_LDS, TENSORcnt double-buffer). All matrix math is
// v_wmma_f32_16x16x32_bf16 with f32 accumulation.
// ---------------------------------------------------------------------------

#define Bc 2
#define Sc 2048
#define Dc 2048
#define Hc 16
#define QLORA 1024
#define KVLORA 512
#define DNOPE 128
#define DROPE 64
#define DV 128
#define DQK 192
#define MTOK (Bc * Sc)          // 4096 tokens
#define EPSf 1e-6f
#define LOG_THETA 9.210340371976184f   // ln(10000)

typedef __attribute__((ext_vector_type(16))) __bf16 v16bf;
typedef __attribute__((ext_vector_type(8)))  float  v8f;
typedef __attribute__((ext_vector_type(4)))  unsigned u32x4;
typedef __attribute__((ext_vector_type(8)))  unsigned u32x8;

union Frag32B { uint4 u[2]; v16bf v; };   // 32 bytes = one A or B fragment

__device__ __forceinline__ unsigned short f2bf(float f) {
    unsigned int u = __float_as_uint(f);
    unsigned int r = (u + 0x7FFFu + ((u >> 16) & 1u)) >> 16;   // RNE
    return (unsigned short)r;
}

__device__ __forceinline__ v8f wmma_bf16(const Frag32B& a, const Frag32B& b, v8f c) {
    return __builtin_amdgcn_wmma_f32_16x16x32_bf16(
        false, a.v, false, b.v, (short)0, c, false, false);
}

// Async DMA: global -> LDS, 16 bytes per lane, tracked by ASYNCcnt.
__device__ __forceinline__ void async_ld_b128(void* lds, const void* g) {
    unsigned           loff = (unsigned)(size_t)lds;
    unsigned long long ga   = (unsigned long long)(size_t)g;
    asm volatile("global_load_async_to_lds_b128 %0, %1, off"
                 :: "v"(loff), "v"(ga) : "memory");
}

// Tensor Data Mover: 2D bf16 tile (tile_d0 x tile_d1 elements, row stride
// stride0 elements) from global -> packed LDS. D# per ISA 8.3/8.4.
// Issue from one wave only; tracked by TENSORcnt; EXEC ignored.
__device__ __forceinline__ void tdm_load_2d(void* lds, const void* g,
                                            unsigned tile_d0, unsigned tile_d1,
                                            unsigned long long stride0) {
    unsigned long long ga = (unsigned long long)(size_t)g;
    u32x4 g0;
    g0[0] = 0x1u;                                       // count=1
    g0[1] = (unsigned)(size_t)lds;                      // lds_addr
    g0[2] = (unsigned)(ga & 0xFFFFFFFFu);               // global_addr[31:0]
    g0[3] = (unsigned)((ga >> 32) & 0x01FFFFFFu) | 0x80000000u; // ga[56:32]|type=2
    u32x8 g1;
    g1[0] = 0x00010000u;                                // data_size=1 (2 bytes)
    g1[1] = (tile_d0 & 0xFFFFu) << 16;                  // tensor_dim0[15:0]
    g1[2] = (tile_d0 >> 16) | ((tile_d1 & 0xFFFFu) << 16); // td0 hi | tensor_dim1 lo
    g1[3] = (tile_d1 >> 16) | ((tile_d0 & 0xFFFFu) << 16); // td1 hi | tile_dim0
    g1[4] = tile_d1 & 0xFFFFu;                          // tile_dim1 (tile_dim2=0)
    g1[5] = (unsigned)(stride0 & 0xFFFFFFFFu);          // tensor_dim0_stride lo
    g1[6] = (unsigned)((stride0 >> 32) & 0xFFFFu);      // stride hi (stride1=0)
    g1[7] = 0u;
    asm volatile("tensor_load_to_lds %0, %1" :: "s"(g0), "s"(g1) : "memory");
}

// ---------------------------------------------------------------------------
// f32 -> bf16 elementwise convert (grid-stride, float4 chunks)
// ---------------------------------------------------------------------------
__global__ __launch_bounds__(256)
void mla_cvt_bf16(const float* __restrict__ src, unsigned short* __restrict__ dst,
                  long n4) {
    for (long i = (long)blockIdx.x * 256 + threadIdx.x; i < n4;
         i += (long)gridDim.x * 256) {
        float4 v = ((const float4*)src)[i];
        unsigned short* d = dst + i * 4;
        d[0] = f2bf(v.x); d[1] = f2bf(v.y); d[2] = f2bf(v.z); d[3] = f2bf(v.w);
    }
}

// ---------------------------------------------------------------------------
// GEMM: C[M,N] = A[M,K] * W[N,K]^T, A/W bf16, C f32.
// Block tile 128x256, K-step 32, 256 thr = 8 waves (4M x 2N), wave = 32x128.
// Tiles staged via async-to-LDS, double buffered, ASYNCcnt pipelined.
// ---------------------------------------------------------------------------
__global__ __launch_bounds__(256)
void mla_gemm_bf(const unsigned short* __restrict__ A,
                 const unsigned short* __restrict__ W,
                 float* __restrict__ C, int N, int K, int lda, int ldc) {
    __shared__ unsigned short sA[2][128][40];   // 80B rows (16B aligned)
    __shared__ unsigned short sB[2][256][40];

    const int tid  = threadIdx.x;
    const int lane = tid & 31;
    const int wave = tid >> 5;
    const int wm   = wave & 3;          // 32-row strip
    const int wn   = wave >> 2;         // 128-col strip
    const int lr   = lane & 15;
    const int lh   = lane >> 4;
    const int mBase = blockIdx.y * 128;
    const int nBase = blockIdx.x * 256;

    // stage a 128x32 A tile + 256x32 B tile (6 async b128 per thread, uniform)
    auto stage = [&](int buf, int k0) {
#pragma unroll
        for (int i = 0; i < 2; ++i) {
            int c = tid + i * 256;
            int r = c >> 2;
            int kc = (c & 3) * 8;
            async_ld_b128(&sA[buf][r][kc],
                          A + (size_t)(mBase + r) * lda + k0 + kc);
        }
#pragma unroll
        for (int i = 0; i < 4; ++i) {
            int c = tid + i * 256;
            int r = c >> 2;
            int kc = (c & 3) * 8;
            int n = nBase + r;
            if (n >= N) n = N - 1;      // clamp: keep issue count uniform
            async_ld_b128(&sB[buf][r][kc],
                          W + (size_t)n * K + k0 + kc);
        }
    };

    v8f acc[2][8];
#pragma unroll
    for (int t = 0; t < 2; ++t)
#pragma unroll
        for (int u = 0; u < 8; ++u)
#pragma unroll
            for (int r = 0; r < 8; ++r) acc[t][u][r] = 0.0f;

    stage(0, 0);
    int buf = 0;
    for (int k0 = 0; k0 < K; k0 += 32) {
        __syncthreads();                    // next buffer free of readers
        if (k0 + 32 < K) {
            stage(buf ^ 1, k0 + 32);
            if (k0 + 64 < K)                // warm L2 one tile further ahead
                __builtin_prefetch(A + (size_t)(mBase + (tid >> 1)) * lda + k0 + 64, 0, 1);
            asm volatile("s_wait_asynccnt 0x6" ::: "memory");
        } else {
            asm volatile("s_wait_asynccnt 0x0" ::: "memory");
        }
        __syncthreads();                    // tile k0 visible to all waves

        Frag32B fa[2];
#pragma unroll
        for (int t = 0; t < 2; ++t) {
            const unsigned short* base = &sA[buf][wm * 32 + t * 16 + lr][lh * 8];
            fa[t].u[0] = *(const uint4*)(base);
            fa[t].u[1] = *(const uint4*)(base + 16);
        }
#pragma unroll
        for (int u = 0; u < 8; ++u) {
            Frag32B fb;
            const unsigned short* base = &sB[buf][wn * 128 + u * 16 + lr][lh * 8];
            fb.u[0] = *(const uint4*)(base);
            fb.u[1] = *(const uint4*)(base + 16);
#pragma unroll
            for (int t = 0; t < 2; ++t)
                acc[t][u] = wmma_bf16(fa[t], fb, acc[t][u]);
        }
        buf ^= 1;
    }

#pragma unroll
    for (int t = 0; t < 2; ++t)
#pragma unroll
        for (int u = 0; u < 8; ++u)
#pragma unroll
            for (int r = 0; r < 8; ++r) {
                int row = mBase + wm * 32 + t * 16 + lh * 8 + r;
                int col = nBase + wn * 128 + u * 16 + lr;
                if (col < N)
                    C[(size_t)row * ldc + col] = acc[t][u][r];
            }
}

// ---------------------------------------------------------------------------
// RMSNorm: read f32 row (stride ld, first `dim` elems), write bf16 row.
// ---------------------------------------------------------------------------
__global__ __launch_bounds__(256)
void mla_rmsnorm_bf(const float* __restrict__ X, const float* __restrict__ w,
                    unsigned short* __restrict__ Y, int ld, int dim) {
    const float* x = X + (size_t)blockIdx.x * ld;
    unsigned short* y = Y + (size_t)blockIdx.x * dim;
    float ss = 0.f;
    for (int i = threadIdx.x; i < dim; i += 256) {
        float v = x[i];
        ss += v * v;
    }
    __shared__ float red[8];
#pragma unroll
    for (int o = 16; o > 0; o >>= 1) ss += __shfl_xor(ss, o, 32);
    if ((threadIdx.x & 31) == 0) red[threadIdx.x >> 5] = ss;
    __syncthreads();
    if (threadIdx.x == 0) {
        float tot = 0.f;
#pragma unroll
        for (int i = 0; i < 8; ++i) tot += red[i];
        red[0] = rsqrtf(tot / (float)dim + EPSf);
    }
    __syncthreads();
    const float rs = red[0];
    for (int i = threadIdx.x; i < dim; i += 256)
        y[i] = f2bf(w[i] * x[i] * rs);
}

// ---------------------------------------------------------------------------
// Q RoPE + layout: q_up [B*S, H, 192] f32 -> qbf [B, H, S, 192] bf16
// ---------------------------------------------------------------------------
__global__ __launch_bounds__(192)
void mla_rope_q(const float* __restrict__ qup, unsigned short* __restrict__ qout) {
    const int tok = blockIdx.x;
    const int h   = blockIdx.y;
    const int j   = threadIdx.x;
    const int b   = tok / Sc;
    const int s   = tok - b * Sc;
    const float* in = qup + ((size_t)tok * Hc + h) * DQK;
    unsigned short* out = qout + (((size_t)b * Hc + h) * Sc + s) * DQK;

    float val;
    if (j < DNOPE) {
        val = in[j];
    } else {
        int jj = j - DNOPE;
        float de, rot;
        if (jj < 32) { de = in[DNOPE + 2 * jj];            rot = -in[DNOPE + 2 * jj + 1]; }
        else         { de = in[DNOPE + 2 * (jj - 32) + 1]; rot =  in[DNOPE + 2 * (jj - 32)]; }
        float invf = __expf(-(float)(jj & 31) * (LOG_THETA / 32.f));
        float ang  = (float)s * invf;
        float sn, cs;
        __sincosf(ang, &sn, &cs);
        val = de * cs + rot * sn;
    }
    out[j] = f2bf(val);
}

// ---------------------------------------------------------------------------
// K/V layout: kv_up [B*S,H,256] f32 + rope(kv_down[:,512:576])
//   -> kbf [B,H,S,192] bf16, vbf [B,H,S,128] bf16 (row-major; transposed later)
// ---------------------------------------------------------------------------
__global__ __launch_bounds__(192)
void mla_rope_kv(const float* __restrict__ kvup, const float* __restrict__ kvdown,
                 unsigned short* __restrict__ kout, unsigned short* __restrict__ vout) {
    const int tok = blockIdx.x;
    const int h   = blockIdx.y;
    const int j   = threadIdx.x;
    const int b   = tok / Sc;
    const int s   = tok - b * Sc;
    const float* up = kvup + ((size_t)tok * Hc + h) * (DNOPE + DV);
    const size_t ob = ((size_t)b * Hc + h) * Sc + s;

    if (j < DNOPE) {
        kout[ob * DQK + j] = f2bf(up[j]);
        vout[ob * DV  + j] = f2bf(up[DNOPE + j]);
    } else {
        int jj = j - DNOPE;
        const float* kr = kvdown + (size_t)tok * (KVLORA + DROPE) + KVLORA;
        float de, rot;
        if (jj < 32) { de = kr[2 * jj];            rot = -kr[2 * jj + 1]; }
        else         { de = kr[2 * (jj - 32) + 1]; rot =  kr[2 * (jj - 32)]; }
        float invf = __expf(-(float)(jj & 31) * (LOG_THETA / 32.f));
        float ang  = (float)s * invf;
        float sn, cs;
        __sincosf(ang, &sn, &cs);
        kout[ob * DQK + j] = f2bf(de * cs + rot * sn);
    }
}

// ---------------------------------------------------------------------------
// V transpose: [B,H,S,128] -> [B,H,128,S] via 32x32 LDS tiles (coalesced).
// ---------------------------------------------------------------------------
__global__ __launch_bounds__(256)
void mla_transpose_v(const unsigned short* __restrict__ src,
                     unsigned short* __restrict__ dst) {
    __shared__ unsigned short tile[32][33];
    const int bh = blockIdx.z;
    const int s0 = blockIdx.x * 32;
    const int d0 = blockIdx.y * 32;
    const unsigned short* in  = src + (size_t)bh * Sc * DV;
    unsigned short*       out = dst + (size_t)bh * DV * Sc;
    const int tid = threadIdx.x;
#pragma unroll
    for (int i = 0; i < 4; ++i) {
        int e = tid + i * 256;
        int r = e >> 5, c = e & 31;
        tile[r][c] = in[(size_t)(s0 + r) * DV + d0 + c];
    }
    __syncthreads();
#pragma unroll
    for (int i = 0; i < 4; ++i) {
        int e = tid + i * 256;
        int r = e >> 5, c = e & 31;
        out[(size_t)(d0 + r) * Sc + s0 + c] = tile[c][r];
    }
}

// ---------------------------------------------------------------------------
// Flash attention: block = (64 q-rows, head, batch), 128 thr = 4 waves, each
// wave owns 16 q rows. 32-key blocks double-buffered via the Tensor Data
// Mover (wave 0 issues, TENSORcnt pipelined). Q/K/Vt bf16; O bf16 [B,S,H*128].
// ---------------------------------------------------------------------------
__global__ __launch_bounds__(128)
void mla_attn(const unsigned short* __restrict__ Q,
              const unsigned short* __restrict__ Kt,
              const unsigned short* __restrict__ Vt,
              unsigned short* __restrict__ O) {
    __shared__ unsigned short sK [2][32][192];   // TDM packs rows: 384B rows
    __shared__ unsigned short sVt[2][128][32];   // 64B rows
    __shared__ unsigned short sP [64][40];       // q rows x 32 keys (padded)

    const int tid  = threadIdx.x;
    const int lane = tid & 31;
    const int wave = tid >> 5;
    const int lr   = lane & 15;
    const int lh   = lane >> 4;
    const int qbase = blockIdx.x * 64;
    const size_t headBase = ((size_t)blockIdx.z * Hc + blockIdx.y) * Sc;
    const unsigned short* qh = Q  + headBase * DQK;
    const unsigned short* kh = Kt + headBase * DQK;
    const unsigned short* vt = Vt + headBase * DV;     // [128][S], row stride S
    const float scale = 0.07216878364870323f;          // 1/sqrt(192)

    // one TDM descriptor pair per tile: K (192x32, stride 192) and
    // Vt (32x128, stride S)
    auto stage = [&](int buf, int kb) {
        tdm_load_2d(&sK[buf][0][0],  kh + (size_t)kb * DQK, DQK, 32, DQK);
        tdm_load_2d(&sVt[buf][0][0], vt + kb,               32,  DV, Sc);
    };

    // Q fragments for this wave's 16 rows (6 K-steps of 32)
    Frag32B fq[6];
    {
        const unsigned short* qrow = qh + (size_t)(qbase + wave * 16 + lr) * DQK;
#pragma unroll
        for (int t = 0; t < 6; ++t) {
            fq[t].u[0] = *(const uint4*)(qrow + t * 32 + lh * 8);
            fq[t].u[1] = *(const uint4*)(qrow + t * 32 + 16 + lh * 8);
        }
    }

    v8f o[8];
#pragma unroll
    for (int n = 0; n < 8; ++n)
#pragma unroll
        for (int r = 0; r < 8; ++r) o[n][r] = 0.0f;
    float m_r[8], l_r[8];
#pragma unroll
    for (int r = 0; r < 8; ++r) { m_r[r] = -1e30f; l_r[r] = 0.f; }

    if (wave == 0) stage(0, 0);
    int buf = 0;
    for (int kb = 0; kb < Sc; kb += 32) {
        __syncthreads();                    // next buffer free of readers
        if (wave == 0) {
            if (kb + 32 < Sc) {
                stage(buf ^ 1, kb + 32);
                asm volatile("s_wait_tensorcnt 0x2" ::: "memory");
            } else {
                asm volatile("s_wait_tensorcnt 0x0" ::: "memory");
            }
        }
        __syncthreads();                    // tile kb visible to all waves

        // scores: 16 q-rows x 32 keys (2 N-tiles x 6 K-steps)
        v8f sc[2];
#pragma unroll
        for (int u = 0; u < 2; ++u)
#pragma unroll
            for (int r = 0; r < 8; ++r) sc[u][r] = 0.0f;
#pragma unroll
        for (int u = 0; u < 2; ++u) {
#pragma unroll
            for (int t = 0; t < 6; ++t) {
                Frag32B fb;
                const unsigned short* base = &sK[buf][u * 16 + lr][t * 32 + lh * 8];
                fb.u[0] = *(const uint4*)(base);
                fb.u[1] = *(const uint4*)(base + 16);
                sc[u] = wmma_bf16(fq[t], fb, sc[u]);
            }
        }

        // online softmax (row r + 8*lh lives across one 16-lane half)
#pragma unroll
        for (int r = 0; r < 8; ++r) {
            float mx = -1e30f;
#pragma unroll
            for (int u = 0; u < 2; ++u) mx = fmaxf(mx, sc[u][r] * scale);
#pragma unroll
            for (int off = 8; off > 0; off >>= 1) mx = fmaxf(mx, __shfl_xor(mx, off, 32));
            float mnew  = fmaxf(m_r[r], mx);
            float alpha = __expf(m_r[r] - mnew);
            m_r[r] = mnew;
            float rs = 0.f;
#pragma unroll
            for (int u = 0; u < 2; ++u) {
                float p = __expf(sc[u][r] * scale - mnew);
                sc[u][r] = p;
                rs += p;
            }
#pragma unroll
            for (int off = 8; off > 0; off >>= 1) rs += __shfl_xor(rs, off, 32);
            l_r[r] = l_r[r] * alpha + rs;
#pragma unroll
            for (int n = 0; n < 8; ++n) o[n][r] *= alpha;
        }

        // P -> LDS (wave-private 16x32 strip; same wave reads it back)
#pragma unroll
        for (int u = 0; u < 2; ++u)
#pragma unroll
            for (int r = 0; r < 8; ++r)
                sP[wave * 16 + lh * 8 + r][u * 16 + lr] = f2bf(sc[u][r]);

        // O += P V (one 32-key WMMA step, 8 dv-tiles)
        {
            Frag32B fa;
            const unsigned short* arow = &sP[wave * 16 + lr][lh * 8];
            fa.u[0] = *(const uint4*)(arow);
            fa.u[1] = *(const uint4*)(arow + 16);
#pragma unroll
            for (int n = 0; n < 8; ++n) {
                Frag32B fb;
                const unsigned short* vb = &sVt[buf][n * 16 + lr][lh * 8];
                fb.u[0] = *(const uint4*)(vb);
                fb.u[1] = *(const uint4*)(vb + 16);
                o[n] = wmma_bf16(fa, fb, o[n]);
            }
        }
        buf ^= 1;
    }

    // epilogue: O /= l ; write bf16 [B, S, H*128]
#pragma unroll
    for (int r = 0; r < 8; ++r) {
        float inv = 1.f / l_r[r];
        int tok = qbase + wave * 16 + lh * 8 + r;
        size_t orow = ((size_t)blockIdx.z * Sc + tok) * (Hc * DV) + (size_t)blockIdx.y * DV;
#pragma unroll
        for (int n = 0; n < 8; ++n)
            O[orow + n * 16 + lr] = f2bf(o[n][r] * inv);
    }
}

// ---------------------------------------------------------------------------
// Workspace layout (bytes)
// ---------------------------------------------------------------------------
static const size_t OFF_HBF   = 0;                                          // hidden bf16
static const size_t OFF_WQD   = OFF_HBF   + (size_t)MTOK * Dc * 2;
static const size_t OFF_WQUP  = OFF_WQD   + (size_t)QLORA * Dc * 2;
static const size_t OFF_WKVD  = OFF_WQUP  + (size_t)(Hc * DQK) * QLORA * 2;
static const size_t OFF_WKVUP = OFF_WKVD  + (size_t)(KVLORA + DROPE) * Dc * 2;
static const size_t OFF_WOUT  = OFF_WKVUP + (size_t)(Hc * (DNOPE + DV)) * KVLORA * 2;
static const size_t OFF_QDOWN = OFF_WOUT  + (size_t)Dc * (Hc * DV) * 2;     // f32
static const size_t OFF_QDBF  = OFF_QDOWN + (size_t)MTOK * QLORA * 4;
static const size_t OFF_QUP   = OFF_QDBF  + (size_t)MTOK * QLORA * 2;       // f32
static const size_t OFF_KVDN  = OFF_QUP   + (size_t)MTOK * (Hc * DQK) * 4;  // f32
static const size_t OFF_CKVBF = OFF_KVDN  + (size_t)MTOK * (KVLORA + DROPE) * 4;
static const size_t OFF_KVUP  = OFF_CKVBF + (size_t)MTOK * KVLORA * 2;      // f32
static const size_t OFF_QBF   = OFF_KVUP  + (size_t)MTOK * (Hc * (DNOPE + DV)) * 4;
static const size_t OFF_KBF   = OFF_QBF   + (size_t)MTOK * Hc * DQK * 2;
static const size_t OFF_VBF   = OFF_KBF   + (size_t)MTOK * Hc * DQK * 2;
static const size_t OFF_VTBF  = OFF_VBF   + (size_t)MTOK * Hc * DV * 2;
static const size_t OFF_ATTNB = OFF_VTBF  + (size_t)MTOK * Hc * DV * 2;

extern "C" void kernel_launch(void* const* d_in, const int* in_sizes, int n_in,
                              void* d_out, int out_size, void* d_ws, size_t ws_size,
                              hipStream_t stream) {
    (void)in_sizes; (void)n_in; (void)out_size; (void)ws_size;
    const float* hidden    = (const float*)d_in[0];
    const float* Wq_down   = (const float*)d_in[1];
    const float* q_norm_w  = (const float*)d_in[2];
    const float* Wq_up     = (const float*)d_in[3];
    const float* Wkv_down  = (const float*)d_in[4];
    const float* kv_norm_w = (const float*)d_in[5];
    const float* Wkv_up    = (const float*)d_in[6];
    const float* Wout      = (const float*)d_in[7];
    float* out = (float*)d_out;

    char* ws = (char*)d_ws;
    unsigned short* hbf   = (unsigned short*)(ws + OFF_HBF);
    unsigned short* wqd   = (unsigned short*)(ws + OFF_WQD);
    unsigned short* wqup  = (unsigned short*)(ws + OFF_WQUP);
    unsigned short* wkvd  = (unsigned short*)(ws + OFF_WKVD);
    unsigned short* wkvup = (unsigned short*)(ws + OFF_WKVUP);
    unsigned short* wout  = (unsigned short*)(ws + OFF_WOUT);
    float*          qdown = (float*)(ws + OFF_QDOWN);
    unsigned short* qdbf  = (unsigned short*)(ws + OFF_QDBF);
    float*          qup   = (float*)(ws + OFF_QUP);
    float*          kvdn  = (float*)(ws + OFF_KVDN);
    unsigned short* ckvbf = (unsigned short*)(ws + OFF_CKVBF);
    float*          kvup  = (float*)(ws + OFF_KVUP);
    unsigned short* qbf   = (unsigned short*)(ws + OFF_QBF);
    unsigned short* kbf   = (unsigned short*)(ws + OFF_KBF);
    unsigned short* vbf   = (unsigned short*)(ws + OFF_VBF);
    unsigned short* vtbf  = (unsigned short*)(ws + OFF_VTBF);
    unsigned short* attnb = (unsigned short*)(ws + OFF_ATTNB);

    // 0. one-time f32 -> bf16 conversions
    mla_cvt_bf16<<<2048, 256, 0, stream>>>(hidden,   hbf,   (long)MTOK * Dc / 4);
    mla_cvt_bf16<<<1024, 256, 0, stream>>>(Wq_down,  wqd,   (long)QLORA * Dc / 4);
    mla_cvt_bf16<<<1024, 256, 0, stream>>>(Wq_up,    wqup,  (long)(Hc * DQK) * QLORA / 4);
    mla_cvt_bf16<<<1024, 256, 0, stream>>>(Wkv_down, wkvd,  (long)(KVLORA + DROPE) * Dc / 4);
    mla_cvt_bf16<<<1024, 256, 0, stream>>>(Wkv_up,   wkvup, (long)(Hc * (DNOPE + DV)) * KVLORA / 4);
    mla_cvt_bf16<<<1024, 256, 0, stream>>>(Wout,     wout,  (long)Dc * (Hc * DV) / 4);

    // 1. q_down = hidden @ Wq_down^T        [4096,1024] f32
    mla_gemm_bf<<<dim3(QLORA / 256, MTOK / 128), 256, 0, stream>>>(
        hbf, wqd, qdown, QLORA, Dc, Dc, QLORA);
    // 2. RMSNorm -> bf16
    mla_rmsnorm_bf<<<MTOK, 256, 0, stream>>>(qdown, q_norm_w, qdbf, QLORA, QLORA);
    // 3. q_up = q_down_n @ Wq_up^T          [4096,3072] f32
    mla_gemm_bf<<<dim3((Hc * DQK) / 256, MTOK / 128), 256, 0, stream>>>(
        qdbf, wqup, qup, Hc * DQK, QLORA, QLORA, Hc * DQK);
    // 4. kv_down = hidden @ Wkv_down^T      [4096,576] f32
    mla_gemm_bf<<<dim3((KVLORA + DROPE + 255) / 256, MTOK / 128), 256, 0, stream>>>(
        hbf, wkvd, kvdn, KVLORA + DROPE, Dc, Dc, KVLORA + DROPE);
    // 5. RMSNorm(kv_down[:, :512]) -> bf16
    mla_rmsnorm_bf<<<MTOK, 256, 0, stream>>>(kvdn, kv_norm_w, ckvbf, KVLORA + DROPE, KVLORA);
    // 6. kv_up = c_kv_n @ Wkv_up^T          [4096,4096] f32
    mla_gemm_bf<<<dim3((Hc * (DNOPE + DV)) / 256, MTOK / 128), 256, 0, stream>>>(
        ckvbf, wkvup, kvup, Hc * (DNOPE + DV), KVLORA, KVLORA, Hc * (DNOPE + DV));
    // 7. RoPE + layout to [B,H,S,d] bf16; transpose V to [B,H,dv,S]
    mla_rope_q <<<dim3(MTOK, Hc), 192, 0, stream>>>(qup, qbf);
    mla_rope_kv<<<dim3(MTOK, Hc), 192, 0, stream>>>(kvup, kvdn, kbf, vbf);
    mla_transpose_v<<<dim3(Sc / 32, DV / 32, Bc * Hc), 256, 0, stream>>>(vbf, vtbf);
    // 8. flash attention -> attn bf16 [B,S,H*128]
    mla_attn<<<dim3(Sc / 64, Hc, Bc), 128, 0, stream>>>(qbf, kbf, vtbf, attnb);
    // 9. out = attn @ Wout^T                [4096,2048] f32
    mla_gemm_bf<<<dim3(Dc / 128, MTOK / 128), 256, 0, stream>>>(
        attnb, wout, out, Dc, Hc * DV, Hc * DV, Dc);
}
